// Conv_16449724743895
// MI455X (gfx1250) — matrix operand order repaired and checked by reading, not compiled, not run on hardware
//
#include <hip/hip_runtime.h>
#include <math.h>

typedef __attribute__((ext_vector_type(16))) _Float16 v16h;
typedef __attribute__((ext_vector_type(8)))  _Float16 v8h;
typedef __attribute__((ext_vector_type(8)))  float    v8f;

#define RES        8
#define NCELL      512               // 8*8*8
#define CH         32
#define GRID_ELEMS (NCELL * CH)      // 16384
#define NTAPS      27
#define WELEMS     (NTAPS * CH * CH) // 27648 per layer

// ---------------------------------------------------------------- zero scratch
__global__ void zero_kernel(float* __restrict__ p, int n) {
    int i = blockIdx.x * blockDim.x + threadIdx.x;
    if (i < n) p[i] = 0.0f;
}

// ---------------------------------------------------------------- scatter
// One wave per point, one channel per lane: 8 wave-wide ds_add_f32 per point
// (channel-major -> 32 lanes hit 32 distinct banks). Per-block private 64KB
// LDS lattice, merged once into global with atomics.
__global__ __launch_bounds__(256) void scatter_kernel(
    const float* __restrict__ pos, const float* __restrict__ feat,
    float* __restrict__ grid0, int N) {
    __shared__ float lat[GRID_ELEMS];
    for (int i = threadIdx.x; i < GRID_ELEMS; i += blockDim.x) lat[i] = 0.0f;
    __syncthreads();

    const int lane = threadIdx.x & 31;
    const int wpb  = blockDim.x >> 5;
    int wid = blockIdx.x * wpb + (threadIdx.x >> 5);
    int nw  = gridDim.x * wpb;

    for (int p = wid; p < N; p += nw) {
        float s0 = pos[3 * p + 0] * (float)RES;
        float s1 = pos[3 * p + 1] * (float)RES;
        float s2 = pos[3 * p + 2] * (float)RES;
        float b0 = floorf(s0), b1 = floorf(s1), b2 = floorf(s2);
        float f0 = s0 - b0, f1 = s1 - b1, f2 = s2 - b2;
        int i0 = ((int)b0) & 7, i1 = ((int)b1) & 7, i2 = ((int)b2) & 7;
        float fv = feat[p * CH + lane];
#pragma unroll
        for (int c = 0; c < 8; ++c) {
            int o0 = (c >> 2) & 1, o1 = (c >> 1) & 1, o2 = c & 1;
            int j0 = (i0 + o0) & 7, j1 = (i1 + o1) & 7, j2 = (i2 + o2) & 7;
            float w = (o0 ? f0 : 1.0f - f0) * (o1 ? f1 : 1.0f - f1) *
                      (o2 ? f2 : 1.0f - f2);
            int cell = (j0 * 8 + j1) * 8 + j2;
            atomicAdd(&lat[cell * CH + lane], w * fv);   // ds_add_f32
        }
    }
    __syncthreads();
    for (int i = threadIdx.x; i < GRID_ELEMS; i += blockDim.x)
        atomicAdd(&grid0[i], lat[i]);
}

// ---------------------------------------------------------------- WMMA conv
__device__ __forceinline__ v16h cat8(v8h lo, v8h hi) {
    return __builtin_shufflevector(lo, hi, 0, 1, 2, 3, 4, 5, 6, 7,
                                   8, 9, 10, 11, 12, 13, 14, 15);
}

// One 16x16 output tile of one periodic-conv layer as a 27-tap chain of
// v_wmma_f32_16x16x32_f16. srcH is the f16 activation grid [512][32] in LDS,
// wt the f16 weights [27][32][32] in LDS. All fragment loads are contiguous
// 16B ds_load_b128s per the documented 16-bit A/B lane layouts.
__device__ __forceinline__ v8f conv_tile(const _Float16* __restrict__ srcH,
                                         const _Float16* __restrict__ wt,
                                         int mTile, int nTile,
                                         int mloc, int hiHalf) {
    int M  = mTile * 16 + mloc;            // this lane's A-row (cell index)
    int c0 = M >> 6, c1 = (M >> 3) & 7, c2 = M & 7;
    v8f acc = {};
#pragma unroll
    for (int k0 = 0; k0 < 3; ++k0) {
        int s0 = (c0 + k0 + 7) & 7;
#pragma unroll
        for (int k1 = 0; k1 < 3; ++k1) {
            int s1 = (c1 + k1 + 7) & 7;
#pragma unroll
            for (int k2 = 0; k2 < 3; ++k2) {
                int s2  = (c2 + k2 + 7) & 7;
                int tap = (k0 * 3 + k1) * 3 + k2;
                // A 16x32 f16 layout: lanes<16 K {0..7,16..23}, lanes>=16 +8
                const _Float16* arow =
                    srcH + ((s0 * 8 + s1) * 8 + s2) * CH + hiHalf * 8;
                v8h alo = *(const v8h*)(arow);        // 16B ds_load_b128
                v8h ahi = *(const v8h*)(arow + 16);
                // B 32x16: lane holds col N=lane%16; lanes<16 K {0..7,16..23}, lanes>=16 +8
                const _Float16* brow =
                    wt + (tap * CH + nTile * 16 + mloc) * CH + hiHalf * 8;
                v8h blo = *(const v8h*)(brow);
                v8h bhi = *(const v8h*)(brow + 16);
                acc = __builtin_amdgcn_wmma_f32_16x16x32_f16(
                    false, cat8(alo, ahi), false, cat8(blo, bhi),
                    (short)0, acc, false, false);
            }
        }
    }
    return acc;
}

// Fused 3-layer pipeline in one workgroup. LDS (~290KB < 320KB/WGP):
//   gF  f32[16384]  residual grid                  64KB
//   gH0 f16[16384]  activation ping                32KB
//   gH1 f16[16384]  activation pong                32KB
//   wt  f16[3*27648] all three layers' weights    162KB
__global__ __launch_bounds__(1024) void conv_kernel(
    const float* __restrict__ grid0, const float* __restrict__ kEnc,
    const float* __restrict__ kInner, const float* __restrict__ kDec,
    float* __restrict__ grid2) {
    extern __shared__ char smem[];
    float*    gF  = (float*)smem;
    _Float16* gH0 = (_Float16*)(gF + GRID_ELEMS);
    _Float16* gH1 = gH0 + GRID_ELEMS;
    _Float16* wt  = gH1 + GRID_ELEMS;
    const int tid    = threadIdx.x;
    const int waveId = tid >> 5;
    const int lane   = tid & 31;
    const int mloc   = lane & 15;
    const int hiHalf = (lane & 16) ? 1 : 0;

    // stage input grid (f16) + all weights (f16) into LDS
    for (int i = tid; i < GRID_ELEMS; i += blockDim.x)
        gH0[i] = (_Float16)grid0[i];
    for (int i = tid; i < WELEMS; i += blockDim.x) {
        wt[i]              = (_Float16)kEnc[i];
        wt[WELEMS + i]     = (_Float16)kInner[i];
        wt[2 * WELEMS + i] = (_Float16)kDec[i];
    }
    __syncthreads();

    // encoder: r = conv(gH0) -> gF (f32) + gH1 (f16)
#pragma unroll 1
    for (int i = 0; i < 2; ++i) {
        int tIdx = waveId * 2 + i, mTile = tIdx >> 1, nTile = tIdx & 1;
        v8f acc = conv_tile(gH0, wt, mTile, nTile, mloc, hiHalf);
        int nBase = nTile * 16 + mloc;
        int mBase = mTile * 16 + hiHalf * 8;
#pragma unroll
        for (int r = 0; r < 8; ++r) {
            int idx = (mBase + r) * CH + nBase;
            gF[idx]  = acc[r];
            gH1[idx] = (_Float16)acc[r];
        }
    }
    __syncthreads();

    // inner residual block: out = r + silu(conv(r)); epilogue fused, reads the
    // same tile of gF this wave wrote above. Writes next activations to gH0.
#pragma unroll 1
    for (int i = 0; i < 2; ++i) {
        int tIdx = waveId * 2 + i, mTile = tIdx >> 1, nTile = tIdx & 1;
        v8f acc = conv_tile(gH1, wt + WELEMS, mTile, nTile, mloc, hiHalf);
        int nBase = nTile * 16 + mloc;
        int mBase = mTile * 16 + hiHalf * 8;
#pragma unroll
        for (int r = 0; r < 8; ++r) {
            int idx = (mBase + r) * CH + nBase;
            float x = acc[r];
            float y = gF[idx] + x / (1.0f + __expf(-x));  // residual + SiLU
            gH0[idx] = (_Float16)y;
        }
    }
    __syncthreads();

    // decoder: conv(gH0) -> global grid2 (f32) directly
#pragma unroll 1
    for (int i = 0; i < 2; ++i) {
        int tIdx = waveId * 2 + i, mTile = tIdx >> 1, nTile = tIdx & 1;
        v8f acc = conv_tile(gH0, wt + 2 * WELEMS, mTile, nTile, mloc, hiHalf);
        int nBase = nTile * 16 + mloc;
        int mBase = mTile * 16 + hiHalf * 8;
#pragma unroll
        for (int r = 0; r < 8; ++r)
            grid2[(mBase + r) * CH + nBase] = acc[r];
    }
}

// ---------------------------------------------------------------- gather
__global__ __launch_bounds__(256) void gather_kernel(
    const float* __restrict__ pos, const float* __restrict__ grid2,
    float* __restrict__ out, int N) {
    __shared__ float g[GRID_ELEMS];
    for (int i = threadIdx.x; i < GRID_ELEMS; i += blockDim.x) g[i] = grid2[i];
    __syncthreads();

    const int lane = threadIdx.x & 31;
    const int wpb  = blockDim.x >> 5;
    int wid = blockIdx.x * wpb + (threadIdx.x >> 5);
    int nw  = gridDim.x * wpb;

    for (int p = wid; p < N; p += nw) {
        float s0 = pos[3 * p + 0] * (float)RES;
        float s1 = pos[3 * p + 1] * (float)RES;
        float s2 = pos[3 * p + 2] * (float)RES;
        int i0 = ((int)floorf(s0)) & 7;
        int i1 = ((int)floorf(s1)) & 7;
        int i2 = ((int)floorf(s2)) & 7;
        float acc = 0.0f;
#pragma unroll
        for (int c = 0; c < 8; ++c) {
            int j0 = (i0 + ((c >> 2) & 1)) & 7;
            int j1 = (i1 + ((c >> 1) & 1)) & 7;
            int j2 = (i2 + (c & 1)) & 7;
            acc += g[((j0 * 8 + j1) * 8 + j2) * CH + lane];
        }
        out[p * CH + lane] = acc;            // coalesced 128B store per wave
    }
}

// ---------------------------------------------------------------- launch
extern "C" void kernel_launch(void* const* d_in, const int* in_sizes, int n_in,
                              void* d_out, int out_size, void* d_ws, size_t ws_size,
                              hipStream_t stream) {
    const float* pos    = (const float*)d_in[0];
    const float* feat   = (const float*)d_in[1];
    const float* kEnc   = (const float*)d_in[2];
    const float* kInner = (const float*)d_in[3];
    const float* kDec   = (const float*)d_in[4];
    float* out = (float*)d_out;
    int N = in_sizes[0] / 3;

    float* grid0 = (float*)d_ws;            // 64KB scatter accumulation grid
    float* grid2 = grid0 + GRID_ELEMS;      // 64KB decoder output grid

    zero_kernel<<<(GRID_ELEMS + 255) / 256, 256, 0, stream>>>(grid0, GRID_ELEMS);
    scatter_kernel<<<120, 256, 0, stream>>>(pos, feat, grid0, N);

    size_t convLds = (size_t)GRID_ELEMS * sizeof(float) +
                     (size_t)(2 * GRID_ELEMS) * sizeof(_Float16) +
                     (size_t)(3 * WELEMS) * sizeof(_Float16);   // ~290KB
    (void)hipFuncSetAttribute((const void*)conv_kernel,
                              hipFuncAttributeMaxDynamicSharedMemorySize,
                              (int)convLds);
    conv_kernel<<<1, 1024, convLds, stream>>>(grid0, kEnc, kInner, kDec, grid2);

    gather_kernel<<<240, 256, 0, stream>>>(pos, grid2, out, N);
}
